// Model_31138512896381
// MI455X (gfx1250) — 8-bit matrix operand order repaired and checked by reading, compiled, not run on hardware
//
#include <hip/hip_runtime.h>
#include <stdint.h>

typedef __attribute__((ext_vector_type(8))) int v8i;

#define MSIZE 4096
#define KSIZE 4096
#define NSIZE 4096

#define LDA 80
#define LDB 80

__global__ __launch_bounds__(256) void pack_kernel(const int* __restrict__ src,
                                                   uint8_t* __restrict__ dst) {
    size_t i = ((size_t)blockIdx.x * 256 + threadIdx.x) * 4;
    int4 v = *(const int4*)(src + i);
    uchar4 o;
    o.x = (uint8_t)v.x; o.y = (uint8_t)v.y; o.z = (uint8_t)v.z; o.w = (uint8_t)v.w;
    *(uchar4*)(dst + i) = o;
}

__global__ __launch_bounds__(256) void pack_transpose_kernel(const int* __restrict__ src,
                                                             uint8_t* __restrict__ dst) {
    __shared__ uint8_t tile[32][36];
    const int t  = threadIdx.x;
    const int r  = t >> 3;
    const int c4 = (t & 7) * 4;
    const int k0 = blockIdx.y * 32;
    const int n0 = blockIdx.x * 32;

    int4 v = *(const int4*)(src + (size_t)(k0 + r) * NSIZE + n0 + c4);
    tile[c4 + 0][r] = (uint8_t)v.x;
    tile[c4 + 1][r] = (uint8_t)v.y;
    tile[c4 + 2][r] = (uint8_t)v.z;
    tile[c4 + 3][r] = (uint8_t)v.w;
    __syncthreads();

    uchar4 o;
    o.x = tile[r][c4 + 0];
    o.y = tile[r][c4 + 1];
    o.z = tile[r][c4 + 2];
    o.w = tile[r][c4 + 3];
    *(uchar4*)(dst + (size_t)(n0 + r) * KSIZE + k0 + c4) = o;
}

__global__ __launch_bounds__(256) void gemm_iu8_kernel(const uint8_t* __restrict__ A,
                                                       const uint8_t* __restrict__ Bt,
                                                       const int* __restrict__ addend,
                                                       float* __restrict__ out) {
    __shared__ uint8_t sA[128 * LDA];
    __shared__ uint8_t sB[128 * LDB];

    const int t    = threadIdx.x;
    const int wave = t >> 5;
    const int lane = t & 31;
    const int wr   = wave >> 1;
    const int wc   = wave & 1;
    const int lrow = lane & 15;
    const int half = lane >> 4;

    const int bm = blockIdx.y * 128;
    const int bn = blockIdx.x * 128;

    const int ldr = t >> 1;
    const int ldo = (t & 1) * 32;
    const uint8_t* gA = A  + (size_t)(bm + ldr) * KSIZE + ldo;
    const uint8_t* gB = Bt + (size_t)(bn + ldr) * KSIZE + ldo;
    uint8_t* sAp = sA + ldr * LDA + ldo;
    uint8_t* sBp = sB + ldr * LDB + ldo;

    v8i acc[2][4] = {};

    for (int k0 = 0; k0 < KSIZE; k0 += 64) {
        uint4 a0 = *(const uint4*)(gA + k0);
        uint4 a1 = *(const uint4*)(gA + k0 + 16);
        uint4 b0 = *(const uint4*)(gB + k0);
        uint4 b1 = *(const uint4*)(gB + k0 + 16);
        if (k0 + 64 < KSIZE) {
            __builtin_prefetch(gA + k0 + 64, 0, 0);
            __builtin_prefetch(gB + k0 + 64, 0, 0);
        }
        __syncthreads();
        *(uint4*)(sAp)      = a0;
        *(uint4*)(sAp + 16) = a1;
        *(uint4*)(sBp)      = b0;
        *(uint4*)(sBp + 16) = b1;
        __syncthreads();

        v8i af[2];
#pragma unroll
        for (int mt = 0; mt < 2; ++mt) {
            const uint8_t* ap = sA + (wr * 32 + mt * 16 + lrow) * LDA + half * 8;
            int2 p0 = *(const int2*)(ap);
            int2 p1 = *(const int2*)(ap + 16);
            int2 p2 = *(const int2*)(ap + 32);
            int2 p3 = *(const int2*)(ap + 48);
            af[mt][0] = p0.x; af[mt][1] = p0.y;
            af[mt][2] = p1.x; af[mt][3] = p1.y;
            af[mt][4] = p2.x; af[mt][5] = p2.y;
            af[mt][6] = p3.x; af[mt][7] = p3.y;
        }

        v8i bf[4];
#pragma unroll
        for (int nt = 0; nt < 4; ++nt) {
            const uint8_t* bp = sB + (wc * 64 + nt * 16 + lrow) * LDB + half * 8;
            int2 q0 = *(const int2*)(bp);
            int2 q1 = *(const int2*)(bp + 16);
            int2 q2 = *(const int2*)(bp + 32);
            int2 q3 = *(const int2*)(bp + 48);
            bf[nt][0] = q0.x; bf[nt][1] = q0.y; bf[nt][2] = q1.x; bf[nt][3] = q1.y;
            bf[nt][4] = q2.x; bf[nt][5] = q2.y; bf[nt][6] = q3.x; bf[nt][7] = q3.y;
        }

#pragma unroll
        for (int mt = 0; mt < 2; ++mt)
#pragma unroll
            for (int nt = 0; nt < 4; ++nt)
                acc[mt][nt] = __builtin_amdgcn_wmma_i32_16x16x64_iu8(
                     false, af[mt],
                     false, bf[nt],
                    acc[mt][nt],
                     false,  false);
    }

#pragma unroll
    for (int mt = 0; mt < 2; ++mt) {
#pragma unroll
        for (int nt = 0; nt < 4; ++nt) {
            const int row0 = bm + wr * 32 + mt * 16 + half * 8;
            const int col  = bn + wc * 64 + nt * 16 + lrow;
#pragma unroll
            for (int r = 0; r < 8; ++r) {
                size_t idx = (size_t)(row0 + r) * NSIZE + col;
                out[idx] = (float)(acc[mt][nt][r] + addend[idx]);
            }
        }
    }
}

extern "C" void kernel_launch(void* const* d_in, const int* in_sizes, int n_in,
                              void* d_out, int out_size, void* d_ws, size_t ws_size,
                              hipStream_t stream) {
    const int* input_tensor = (const int*)d_in[0];
    const int* mat1         = (const int*)d_in[1];
    const int* mat2         = (const int*)d_in[2];
    float* out = (float*)d_out;

    uint8_t* Ap = (uint8_t*)d_ws;
    uint8_t* Bt = (uint8_t*)d_ws + (size_t)MSIZE * KSIZE;

    pack_kernel<<<(size_t)MSIZE * KSIZE / (256 * 4), 256, 0, stream>>>(mat1, Ap);

    dim3 tg(NSIZE / 32, KSIZE / 32);
    pack_transpose_kernel<<<tg, 256, 0, stream>>>(mat2, Bt);

    dim3 grid(NSIZE / 128, MSIZE / 128);
    gemm_iu8_kernel<<<grid, 256, 0, stream>>>(Ap, Bt, input_tensor, out);
}
